// MetabolicGNN_39041252720910
// MI455X (gfx1250) — matrix operand order repaired and checked by reading, not compiled, not run on hardware
//
#include <hip/hip_runtime.h>

// ---------------------------------------------------------------------------
// MetabolicGNN for MI455X (gfx1250): bf16 WMMA GEMMs + vectorized edge kernels
// ---------------------------------------------------------------------------

typedef __attribute__((ext_vector_type(16))) __bf16 v16bf;
typedef __attribute__((ext_vector_type(8)))  __bf16 v8bf;
typedef __attribute__((ext_vector_type(8)))  float  v8f;

constexpr int IN_DIM  = 256;
constexpr int HID     = 128;
constexpr int HEADS   = 4;
constexpr int HO      = 512;   // HEADS * HID
constexpr int OUT_DIM = 64;

// ----- helpers --------------------------------------------------------------

__device__ __forceinline__ unsigned fmap(float f) {
  unsigned u = __float_as_uint(f);
  return (u & 0x80000000u) ? ~u : (u | 0x80000000u);
}
__device__ __forceinline__ float funmap(unsigned u) {
  return __uint_as_float((u & 0x80000000u) ? (u & 0x7fffffffu) : ~u);
}
__device__ __forceinline__ float lrelu(float x) { return x > 0.f ? x : 0.2f * x; }

// ----- fp32 -> bf16 convert (plain, for activations) ------------------------

__global__ void k_f32_to_bf16(const float* __restrict__ in,
                              __bf16* __restrict__ out, int n) {
  int i = blockIdx.x * blockDim.x + threadIdx.x;
  if (i < n) out[i] = (__bf16)in[i];
}

// ----- fp32 [K,Nc] -> bf16 transposed [Nc,K] (for weights) ------------------
// Lets every lane of the GEMM fetch its 16 B-values as one contiguous 32B load.

__global__ void k_f32_to_bf16_T(const float* __restrict__ in,
                                __bf16* __restrict__ out, int K, int Nc) {
  int i = blockIdx.x * blockDim.x + threadIdx.x;
  if (i >= K * Nc) return;
  int k = i / Nc, n = i - k * Nc;
  out[(size_t)n * K + k] = (__bf16)in[i];
}

// ----- bf16 WMMA GEMM: C[M,Nc] = A[M,K] @ B[K,Nc] (+bias)(+relu) ------------
// Bt is B transposed ([Nc,K]). One wave per 16x64 output tile: the A fragment
// is reused across 4 v_wmma_f32_16x16x32_bf16 issues per k-step.

template <int K>
__global__ void k_gemm_bf16_wmma(const __bf16* __restrict__ A,
                                 const __bf16* __restrict__ Bt,
                                 const float* __restrict__ bias,
                                 float* __restrict__ Cf,
                                 __bf16* __restrict__ Cbf,
                                 int M, int Nc, int relu) {
  const int wid    = threadIdx.x >> 5;
  const int lane   = threadIdx.x & 31;
  const int tile   = blockIdx.x * (blockDim.x >> 5) + wid;
  const int ngrp   = Nc >> 6;                       // groups of 64 columns
  const int total  = (M >> 4) * ngrp;
  if (tile >= total) return;                        // wave-uniform guard
  const int tm = tile / ngrp;
  const int tn = tile - tm * ngrp;

  const int half = lane >> 4;                       // 0: lanes 0-15, 1: 16-31
  const int l16  = lane & 15;
  const int arow = tm * 16 + l16;                   // A row held by this lane
  const int cb   = tn * 64;                         // first column of the tile

  v8f acc0 = {}, acc1 = {}, acc2 = {}, acc3 = {};

  // bias per column sub-tile, loaded once
  float bv0 = 0.f, bv1 = 0.f, bv2 = 0.f, bv3 = 0.f;
  if (bias) {
    bv0 = bias[cb + 0 * 16 + l16];
    bv1 = bias[cb + 1 * 16 + l16];
    bv2 = bias[cb + 2 * 16 + l16];
    bv3 = bias[cb + 3 * 16 + l16];
  }

#pragma unroll 4
  for (int k0 = 0; k0 < K; k0 += 32) {
    // A fragment (16x32): half 0 holds K 0..15, half 1 K 16..31
    const v8bf a_lo = *(const v8bf*)(A + (size_t)arow * K + (k0 + half * 16));
    const v8bf a_hi = *(const v8bf*)(A + (size_t)arow * K + (k0 + half * 16 + 8));
    const v16bf afrag = __builtin_shufflevector(
        a_lo, a_hi, 0, 1, 2, 3, 4, 5, 6, 7, 8, 9, 10, 11, 12, 13, 14, 15);

    // B fragments (32x16 each): lanes 0-15 carry K=0..15, lanes 16-31 K=16..31;
    // contiguous in Bt -> single 32B load per sub-tile.
    const v16bf b0 = *(const v16bf*)(Bt + (size_t)(cb + 0 * 16 + l16) * K + k0 + half * 16);
    const v16bf b1 = *(const v16bf*)(Bt + (size_t)(cb + 1 * 16 + l16) * K + k0 + half * 16);
    const v16bf b2 = *(const v16bf*)(Bt + (size_t)(cb + 2 * 16 + l16) * K + k0 + half * 16);
    const v16bf b3 = *(const v16bf*)(Bt + (size_t)(cb + 3 * 16 + l16) * K + k0 + half * 16);

    acc0 = __builtin_amdgcn_wmma_f32_16x16x32_bf16(false, afrag, false, b0, (short)0, acc0, false, false);
    acc1 = __builtin_amdgcn_wmma_f32_16x16x32_bf16(false, afrag, false, b1, (short)0, acc1, false, false);
    acc2 = __builtin_amdgcn_wmma_f32_16x16x32_bf16(false, afrag, false, b2, (short)0, acc2, false, false);
    acc3 = __builtin_amdgcn_wmma_f32_16x16x32_bf16(false, afrag, false, b3, (short)0, acc3, false, false);
  }

  // epilogue: bias + relu, then scalar-branched store loops (hoisted tests)
  float vals[4][8];
#pragma unroll
  for (int r = 0; r < 8; ++r) {
    vals[0][r] = acc0[r] + bv0;
    vals[1][r] = acc1[r] + bv1;
    vals[2][r] = acc2[r] + bv2;
    vals[3][r] = acc3[r] + bv3;
  }
  if (relu) {
#pragma unroll
    for (int q = 0; q < 4; ++q)
#pragma unroll
      for (int r = 0; r < 8; ++r) vals[q][r] = fmaxf(vals[q][r], 0.0f);
  }
  // C/D layout: VGPR r -> row (half*8 + r), column = l16 within 16-wide tile
  if (Cf) {
#pragma unroll
    for (int q = 0; q < 4; ++q) {
      const int col = cb + q * 16 + l16;
#pragma unroll
      for (int r = 0; r < 8; ++r)
        Cf[(size_t)(tm * 16 + half * 8 + r) * Nc + col] = vals[q][r];
    }
  }
  if (Cbf) {
#pragma unroll
    for (int q = 0; q < 4; ++q) {
      const int col = cb + q * 16 + l16;
#pragma unroll
      for (int r = 0; r < 8; ++r)
        Cbf[(size_t)(tm * 16 + half * 8 + r) * Nc + col] = (__bf16)vals[q][r];
    }
  }
}

// ----- GCN: degree / norm ---------------------------------------------------

__global__ void k_deg(const int* __restrict__ ei, float* __restrict__ deg, int E) {
  int e = blockIdx.x * blockDim.x + threadIdx.x;
  if (e < E) atomicAdd(&deg[ei[E + e]], 1.0f);
}

__global__ void k_dinv(float* __restrict__ deg, int Nn) {
  int i = blockIdx.x * blockDim.x + threadIdx.x;
  if (i < Nn) deg[i] = rsqrtf(deg[i] + 1.0f);   // +1 self loop; deg>=1 always
}

// ----- GCN: self-loop init + edge aggregation -------------------------------

__global__ void k_gcn_self(const float* __restrict__ xw, const float* __restrict__ dinv,
                           float* __restrict__ agg, int Nn) {
  int gid = blockIdx.x * blockDim.x + threadIdx.x;
  if (gid >= Nn * 32) return;
  int n = gid >> 5, c = (gid & 31) * 4;
  float s = dinv[n]; s *= s;
  float4 v = *(const float4*)(xw + (size_t)n * HID + c);
  v.x *= s; v.y *= s; v.z *= s; v.w *= s;
  *(float4*)(agg + (size_t)n * HID + c) = v;
}

__global__ void k_gcn_edge(const int* __restrict__ ei, const float* __restrict__ xw,
                           const float* __restrict__ dinv, float* __restrict__ agg,
                           int E) {
  int gid = blockIdx.x * blockDim.x + threadIdx.x;
  int e = gid >> 5;
  if (e >= E) return;
  int lane = threadIdx.x & 31;
  int src = ei[e], dst = ei[E + e];
  if (e + 64 < E)                      // global_prefetch hint for a future row
    __builtin_prefetch(xw + (size_t)ei[e + 64] * HID + lane * 4, 0, 1);
  float nrm = dinv[src] * dinv[dst];
  float4 v = *(const float4*)(xw + (size_t)src * HID + lane * 4);
  float* o = agg + (size_t)dst * HID + lane * 4;
  atomicAdd(o + 0, v.x * nrm);
  atomicAdd(o + 1, v.y * nrm);
  atomicAdd(o + 2, v.z * nrm);
  atomicAdd(o + 3, v.w * nrm);
}

// ----- fused bias + LayerNorm + ReLU (+residual); wave per row --------------

__global__ void k_ln_relu(const float* __restrict__ agg, const float* __restrict__ bias,
                          const float* __restrict__ gamma, const float* __restrict__ beta,
                          const float* __restrict__ resid,
                          float* __restrict__ outf, __bf16* __restrict__ outbf, int Nn) {
  int row = (blockIdx.x * blockDim.x + threadIdx.x) >> 5;
  if (row >= Nn) return;
  int lane = threadIdx.x & 31, c = lane * 4;
  float4 h = *(const float4*)(agg + (size_t)row * HID + c);
  float4 b = *(const float4*)(bias + c);
  float x0 = h.x + b.x, x1 = h.y + b.y, x2 = h.z + b.z, x3 = h.w + b.w;
  float s = x0 + x1 + x2 + x3;
#pragma unroll
  for (int o = 16; o > 0; o >>= 1) s += __shfl_xor(s, o, 32);
  float mu = s * (1.0f / HID);
  float d0 = x0 - mu, d1 = x1 - mu, d2 = x2 - mu, d3 = x3 - mu;
  float ss = d0 * d0 + d1 * d1 + d2 * d2 + d3 * d3;
#pragma unroll
  for (int o = 16; o > 0; o >>= 1) ss += __shfl_xor(ss, o, 32);
  float inv = rsqrtf(ss * (1.0f / HID) + 1e-5f);
  float4 g  = *(const float4*)(gamma + c);
  float4 be = *(const float4*)(beta + c);
  float y0 = fmaxf(d0 * inv * g.x + be.x, 0.f);
  float y1 = fmaxf(d1 * inv * g.y + be.y, 0.f);
  float y2 = fmaxf(d2 * inv * g.z + be.z, 0.f);
  float y3 = fmaxf(d3 * inv * g.w + be.w, 0.f);
  if (resid) {
    float4 r = *(const float4*)(resid + (size_t)row * HID + c);
    y0 += r.x; y1 += r.y; y2 += r.z; y3 += r.w;
  }
  if (outf) *(float4*)(outf + (size_t)row * HID + c) = make_float4(y0, y1, y2, y3);
  if (outbf) {
    __bf16* p = outbf + (size_t)row * HID + c;
    p[0] = (__bf16)y0; p[1] = (__bf16)y1; p[2] = (__bf16)y2; p[3] = (__bf16)y3;
  }
}

// ----- GAT ------------------------------------------------------------------

__global__ void k_gat_scores(const float* __restrict__ xwG,
                             const float* __restrict__ att_src,
                             const float* __restrict__ att_dst,
                             float* __restrict__ a_s, float* __restrict__ a_d, int Nn) {
  int idx = (blockIdx.x * blockDim.x + threadIdx.x) >> 5;
  if (idx >= Nn * HEADS) return;
  int n = idx >> 2, h = idx & 3;
  int lane = threadIdx.x & 31, c = lane * 4;
  float4 v  = *(const float4*)(xwG + (size_t)n * HO + h * HID + c);
  float4 as = *(const float4*)(att_src + h * HID + c);
  float4 ad = *(const float4*)(att_dst + h * HID + c);
  float s1 = v.x * as.x + v.y * as.y + v.z * as.z + v.w * as.w;
  float s2 = v.x * ad.x + v.y * ad.y + v.z * ad.z + v.w * ad.w;
#pragma unroll
  for (int o = 16; o > 0; o >>= 1) { s1 += __shfl_xor(s1, o, 32); s2 += __shfl_xor(s2, o, 32); }
  if (lane == 0) { a_s[idx] = s1; a_d[idx] = s2; }
}

__global__ void k_gat_max_init(const float* __restrict__ a_s, const float* __restrict__ a_d,
                               unsigned* __restrict__ mm, int Nn) {
  int idx = blockIdx.x * blockDim.x + threadIdx.x;
  if (idx < Nn * HEADS) mm[idx] = fmap(lrelu(a_s[idx] + a_d[idx]));
}

__global__ void k_gat_edge_max(const int* __restrict__ ei,
                               const float* __restrict__ a_s, const float* __restrict__ a_d,
                               unsigned* __restrict__ mm, int E) {
  int e = blockIdx.x * blockDim.x + threadIdx.x;
  if (e >= E) return;
  int src = ei[e], dst = ei[E + e];
#pragma unroll
  for (int h = 0; h < HEADS; ++h) {
    float v = lrelu(a_s[src * HEADS + h] + a_d[dst * HEADS + h]);
    atomicMax(&mm[dst * HEADS + h], fmap(v));
  }
}

__global__ void k_gat_m_final(const float* __restrict__ a_s, const float* __restrict__ a_d,
                              unsigned* __restrict__ mm, float* __restrict__ denom, int Nn) {
  int idx = blockIdx.x * blockDim.x + threadIdx.x;
  if (idx >= Nn * HEADS) return;
  float m = funmap(mm[idx]);
  ((float*)mm)[idx] = m;
  float es = lrelu(a_s[idx] + a_d[idx]);
  denom[idx] = expf(es - m);
}

__global__ void k_gat_edge_exp(const int* __restrict__ ei,
                               const float* __restrict__ a_s, const float* __restrict__ a_d,
                               const float* __restrict__ m, float* __restrict__ denom,
                               float* __restrict__ exe, int E) {
  int e = blockIdx.x * blockDim.x + threadIdx.x;
  if (e >= E) return;
  int src = ei[e], dst = ei[E + e];
#pragma unroll
  for (int h = 0; h < HEADS; ++h) {
    float v  = lrelu(a_s[src * HEADS + h] + a_d[dst * HEADS + h]);
    float ex = expf(v - m[dst * HEADS + h]);
    exe[(size_t)e * HEADS + h] = ex;
    atomicAdd(&denom[dst * HEADS + h], ex);
  }
}

__global__ void k_gat_self_msg(const float* __restrict__ xwG,
                               const float* __restrict__ a_s, const float* __restrict__ a_d,
                               const float* __restrict__ m, const float* __restrict__ denom,
                               float* __restrict__ out) {
  int n = blockIdx.x, t = threadIdx.x;
  int h = t >> 5, c = (t & 31) * 4;
  int idx = n * HEADS + h;
  float es = lrelu(a_s[idx] + a_d[idx]);
  float alpha = expf(es - m[idx]) / denom[idx];
  float4 v = *(const float4*)(xwG + (size_t)n * HO + h * HID + c);
  v.x *= alpha; v.y *= alpha; v.z *= alpha; v.w *= alpha;
  *(float4*)(out + (size_t)n * HO + h * HID + c) = v;
}

__global__ void k_gat_edge_msg(const int* __restrict__ ei, const float* __restrict__ xwG,
                               const float* __restrict__ exe, const float* __restrict__ denom,
                               float* __restrict__ out, int E) {
  int e = blockIdx.x, t = threadIdx.x;
  int src = ei[e], dst = ei[E + e];
  int h = t >> 5, c = (t & 31) * 4;
  float alpha = exe[(size_t)e * HEADS + h] / denom[dst * HEADS + h];
  float4 v = *(const float4*)(xwG + (size_t)src * HO + h * HID + c);
  float* o = out + (size_t)dst * HO + h * HID + c;
  atomicAdd(o + 0, v.x * alpha);
  atomicAdd(o + 1, v.y * alpha);
  atomicAdd(o + 2, v.z * alpha);
  atomicAdd(o + 3, v.w * alpha);
}

__global__ void k_gat_final(const float* __restrict__ out, const float* __restrict__ bgat,
                            __bf16* __restrict__ abf, int n) {
  int i = blockIdx.x * blockDim.x + threadIdx.x;
  if (i < n) abf[i] = (__bf16)(out[i] + bgat[i & (HO - 1)]);
}

// ---------------------------------------------------------------------------

extern "C" void kernel_launch(void* const* d_in, const int* in_sizes, int n_in,
                              void* d_out, int out_size, void* d_ws, size_t ws_size,
                              hipStream_t stream) {
  const float* x      = (const float*)d_in[0];
  const int*   ei     = (const int*)  d_in[1];
  const float* Win    = (const float*)d_in[2];
  const float* bin_   = (const float*)d_in[3];
  const float* Wg1    = (const float*)d_in[4];
  const float* bg1    = (const float*)d_in[5];
  const float* g1g    = (const float*)d_in[6];
  const float* g1b    = (const float*)d_in[7];
  const float* Wg2    = (const float*)d_in[8];
  const float* bg2    = (const float*)d_in[9];
  const float* g2g    = (const float*)d_in[10];
  const float* g2b    = (const float*)d_in[11];
  const float* Wgat   = (const float*)d_in[12];
  const float* attS   = (const float*)d_in[13];
  const float* attD   = (const float*)d_in[14];
  const float* bgat   = (const float*)d_in[15];
  const float* Watt   = (const float*)d_in[16];
  const float* batt   = (const float*)d_in[17];
  const float* Wout   = (const float*)d_in[18];
  const float* bout   = (const float*)d_in[19];

  const int Nn = in_sizes[0] / IN_DIM;     // 50000
  const int E  = in_sizes[1] / 2;          // 800000

  // ---- workspace carve-out ----
  char* ws = (char*)d_ws;
  size_t off = 0;
  auto carve = [&](size_t bytes) -> char* {
    char* p = ws + off;
    off = (off + bytes + 255) & ~(size_t)255;
    return p;
  };
  __bf16* x_bf    = (__bf16*)carve((size_t)Nn * IN_DIM * 2);
  __bf16* WinT    = (__bf16*)carve((size_t)IN_DIM * HID * 2);   // [HID, IN_DIM]
  __bf16* Wg1T    = (__bf16*)carve((size_t)HID * HID * 2);
  __bf16* Wg2T    = (__bf16*)carve((size_t)HID * HID * 2);
  __bf16* WgatT   = (__bf16*)carve((size_t)HID * HO * 2);       // [HO, HID]
  __bf16* WattT   = (__bf16*)carve((size_t)HO * HID * 2);       // [HID, HO]
  __bf16* WoutT   = (__bf16*)carve((size_t)HID * OUT_DIM * 2);  // [OUT, HID]
  __bf16* h0_bf   = (__bf16*)carve((size_t)Nn * HID * 2);
  float*  xw      = (float*) carve((size_t)Nn * HID * 4);   // reused both GCN layers
  float*  dinv    = (float*) carve((size_t)Nn * 4);
  float*  agg     = (float*) carve((size_t)Nn * HID * 4);   // reused both GCN layers
  float*  x1_f    = (float*) carve((size_t)Nn * HID * 4);
  __bf16* x1_bf   = (__bf16*)carve((size_t)Nn * HID * 2);
  __bf16* x2_bf   = (__bf16*)carve((size_t)Nn * HID * 2);
  float*  xwG     = (float*) carve((size_t)Nn * HO * 4);
  float*  a_s     = (float*) carve((size_t)Nn * HEADS * 4);
  float*  a_d     = (float*) carve((size_t)Nn * HEADS * 4);
  unsigned* mm    = (unsigned*)carve((size_t)Nn * HEADS * 4);  // mapped max -> float m
  float*  denom   = (float*) carve((size_t)Nn * HEADS * 4);
  float*  exe     = (float*) carve((size_t)E * HEADS * 4);
  float*  gat_out = (float*) carve((size_t)Nn * HO * 4);
  __bf16* a_bf    = (__bf16*)carve((size_t)Nn * HO * 2);
  __bf16* a2_bf   = (__bf16*)carve((size_t)Nn * HID * 2);

  auto cdiv = [](long a, long b) { return (unsigned)((a + b - 1) / b); };
  auto conv = [&](const float* src, __bf16* dst, int n) {
    k_f32_to_bf16<<<cdiv(n, 256), 256, 0, stream>>>(src, dst, n);
  };
  auto convT = [&](const float* src, __bf16* dst, int K, int Nc) {
    k_f32_to_bf16_T<<<cdiv((long)K * Nc, 256), 256, 0, stream>>>(src, dst, K, Nc);
  };

  // ---- bf16 staging (weights transposed for contiguous B fragments) ----
  conv (x,    x_bf,  Nn * IN_DIM);
  convT(Win,  WinT,  IN_DIM, HID);
  convT(Wg1,  Wg1T,  HID, HID);
  convT(Wg2,  Wg2T,  HID, HID);
  convT(Wgat, WgatT, HID, HO);
  convT(Watt, WattT, HO, HID);
  convT(Wout, WoutT, HID, OUT_DIM);

  // ---- h0 = relu(x @ Win + bin) ----
  k_gemm_bf16_wmma<IN_DIM><<<cdiv((long)(Nn / 16) * (HID / 64), 4), 128, 0, stream>>>(
      x_bf, WinT, bin_, nullptr, h0_bf, Nn, HID, 1);

  // ---- symmetric GCN normalization (same for both layers) ----
  hipMemsetAsync(dinv, 0, (size_t)Nn * 4, stream);
  k_deg <<<cdiv(E, 256), 256, 0, stream>>>(ei, dinv, E);
  k_dinv<<<cdiv(Nn, 256), 256, 0, stream>>>(dinv, Nn);

  // ---- GCN layer 0 ----
  k_gemm_bf16_wmma<HID><<<cdiv((long)(Nn / 16) * (HID / 64), 4), 128, 0, stream>>>(
      h0_bf, Wg1T, nullptr, xw, nullptr, Nn, HID, 0);
  k_gcn_self<<<cdiv((long)Nn * 32, 256), 256, 0, stream>>>(xw, dinv, agg, Nn);
  k_gcn_edge<<<cdiv((long)E * 32, 256), 256, 0, stream>>>(ei, xw, dinv, agg, E);
  k_ln_relu <<<cdiv((long)Nn * 32, 256), 256, 0, stream>>>(agg, bg1, g1g, g1b, nullptr,
                                                           x1_f, x1_bf, Nn);

  // ---- GCN layer 1 (residual) ----
  k_gemm_bf16_wmma<HID><<<cdiv((long)(Nn / 16) * (HID / 64), 4), 128, 0, stream>>>(
      x1_bf, Wg2T, nullptr, xw, nullptr, Nn, HID, 0);
  k_gcn_self<<<cdiv((long)Nn * 32, 256), 256, 0, stream>>>(xw, dinv, agg, Nn);
  k_gcn_edge<<<cdiv((long)E * 32, 256), 256, 0, stream>>>(ei, xw, dinv, agg, E);
  k_ln_relu <<<cdiv((long)Nn * 32, 256), 256, 0, stream>>>(agg, bg2, g2g, g2b, x1_f,
                                                           nullptr, x2_bf, Nn);

  // ---- GAT ----
  k_gemm_bf16_wmma<HID><<<cdiv((long)(Nn / 16) * (HO / 64), 4), 128, 0, stream>>>(
      x2_bf, WgatT, nullptr, xwG, nullptr, Nn, HO, 0);
  k_gat_scores  <<<cdiv((long)Nn * HEADS * 32, 256), 256, 0, stream>>>(xwG, attS, attD, a_s, a_d, Nn);
  k_gat_max_init<<<cdiv((long)Nn * HEADS, 256), 256, 0, stream>>>(a_s, a_d, mm, Nn);
  k_gat_edge_max<<<cdiv(E, 256), 256, 0, stream>>>(ei, a_s, a_d, mm, E);
  k_gat_m_final <<<cdiv((long)Nn * HEADS, 256), 256, 0, stream>>>(a_s, a_d, mm, denom, Nn);
  k_gat_edge_exp<<<cdiv(E, 256), 256, 0, stream>>>(ei, a_s, a_d, (const float*)mm, denom, exe, E);
  k_gat_self_msg<<<Nn, 128, 0, stream>>>(xwG, a_s, a_d, (const float*)mm, denom, gat_out);
  k_gat_edge_msg<<<E, 128, 0, stream>>>(ei, xwG, exe, denom, gat_out, E);
  k_gat_final   <<<cdiv((long)Nn * HO, 256), 256, 0, stream>>>(gat_out, bgat, a_bf, Nn * HO);

  // ---- a2 = relu(a @ Watt + batt); out = a2 @ Wout + bout ----
  k_gemm_bf16_wmma<HO><<<cdiv((long)(Nn / 16) * (HID / 64), 4), 128, 0, stream>>>(
      a_bf, WattT, batt, nullptr, a2_bf, Nn, HID, 1);
  k_gemm_bf16_wmma<HID><<<cdiv((long)(Nn / 16) * (OUT_DIM / 64), 4), 128, 0, stream>>>(
      a2_bf, WoutT, bout, (float*)d_out, nullptr, Nn, OUT_DIM, 0);

  (void)n_in; (void)out_size; (void)ws_size;
}
